// MVN_DDI_Block_4483945857415
// MI455X (gfx1250) — matrix operand order repaired and checked by reading, not compiled, not run on hardware
//
#include <hip/hip_runtime.h>

typedef __attribute__((ext_vector_type(16))) _Float16 v16h;
typedef __attribute__((ext_vector_type(8)))  _Float16 v8h;
typedef __attribute__((ext_vector_type(8)))  float    v8f;

// ---------------------------------------------------------------------------
// utility kernels
// ---------------------------------------------------------------------------
__global__ void k_fill_zero(float* p, long long n) {
  long long i = (long long)blockIdx.x * blockDim.x + threadIdx.x;
  long long st = (long long)gridDim.x * blockDim.x;
  for (; i < n; i += st) p[i] = 0.0f;
}

__global__ void k_cvt_f16(const float* __restrict__ x, _Float16* __restrict__ y, long long n) {
  long long i = (long long)blockIdx.x * blockDim.x + threadIdx.x;
  long long st = (long long)gridDim.x * blockDim.x;
  for (; i < n; i += st) y[i] = (_Float16)x[i];
}

// ---------------------------------------------------------------------------
// WMMA GEMM:  Y[M,NOUT] = X[M,K] @ W[NOUT,K]^T  (+bias, epilogue variants)
// A/B in f16, accumulate f32 via v_wmma_f32_16x16x32_f16. One wave = 16-row
// strip (A frags loaded once, reused across NOUT/16 column tiles).
// EPI 0: store f32.  EPI 1: store f32 to Y and relu to Y2.  EPI 2: atomicAdd(relu).
// ---------------------------------------------------------------------------
template<int K, int NOUT, int EPI>
__global__ __launch_bounds__(256) void k_gemm_wmma(
    const _Float16* __restrict__ X, const _Float16* __restrict__ Wt,
    const float* __restrict__ bias, float* __restrict__ Y,
    float* __restrict__ Y2, int M)
{
  constexpr int KT = K / 32;
  constexpr int NT = NOUT / 16;
  const int wave = (int)(((unsigned)blockIdx.x * blockDim.x + threadIdx.x) >> 5);
  const int lane = threadIdx.x & 31;
  const int row0 = wave * 16;
  if (row0 >= M) return;                      // wave-uniform: EXEC stays all-ones

  // A fragment: lane<16 -> row M=lane, K {kbase..kbase+7} and {kbase+16..kbase+23}
  const int mrow  = row0 + (lane & 15);
  const int kbase = (lane & 16) ? 8 : 0;
  v16h afrag[KT];
  const _Float16* xr = X + (size_t)mrow * K + kbase;
#pragma unroll
  for (int kt = 0; kt < KT; ++kt) {
    v8h lo = *(const v8h*)(xr + kt * 32);
    v8h hi = *(const v8h*)(xr + kt * 32 + 16);
    afrag[kt] = __builtin_shufflevector(lo, hi, 0,1,2,3,4,5,6,7,8,9,10,11,12,13,14,15);
  }

  const int ncol  = lane & 15;
  const int khalf = (lane & 16) ? 8 : 0;
  const int rbase = row0 + ((lane & 16) ? 8 : 0);

#pragma unroll
  for (int nt = 0; nt < NT; ++nt) {
    v8f acc = {};
    const _Float16* wr = Wt + (size_t)(nt * 16 + ncol) * K + khalf;
#pragma unroll
    for (int kt = 0; kt < KT; ++kt) {
      v8h b0 = *(const v8h*)(wr + kt * 32);
      v8h b1 = *(const v8h*)(wr + kt * 32 + 16);
      v16h bfrag = __builtin_shufflevector(b0, b1, 0,1,2,3,4,5,6,7,8,9,10,11,12,13,14,15);
      acc = __builtin_amdgcn_wmma_f32_16x16x32_f16(false, afrag[kt], false, bfrag,
                                                   (short)0, acc, false, false);
    }
    const int n = nt * 16 + ncol;
    const float bv = bias ? bias[n] : 0.0f;
#pragma unroll
    for (int v = 0; v < 8; ++v) {
      const int r = rbase + v;
      float val = acc[v] + bv;
      size_t o = (size_t)r * NOUT + n;
      if constexpr (EPI == 0) {
        Y[o] = val;
      } else if constexpr (EPI == 1) {
        Y[o] = val;
        Y2[o] = val > 0.0f ? val : 0.0f;
      } else {
        atomicAdd(&Y[o], val > 0.0f ? val : 0.0f);
      }
    }
  }
}

// ---------------------------------------------------------------------------
// column sums of [M,64] (for edge_attr mean)
// ---------------------------------------------------------------------------
__global__ void k_colsum64(const float* __restrict__ x, long long M, float* __restrict__ cs) {
  const int c = threadIdx.x;                    // 64 threads
  long long r0 = (long long)blockIdx.x * 2048;
  long long r1 = r0 + 2048; if (r1 > M) r1 = M;
  float s = 0.0f;
  for (long long r = r0; r < r1; ++r) s += x[r * 64 + c];
  atomicAdd(&cs[c], s);
}

// q[h,k] = sum_c W_edge[h*64+c, k] * att_edge[h,c]; loop consts = mean . q
__global__ void k_qprep(const float* __restrict__ Wedge, const float* __restrict__ att,
                        const float* __restrict__ csh, const float* __restrict__ cst,
                        float Einv, float* __restrict__ q,
                        float* __restrict__ loop_h, float* __restrict__ loop_t) {
  int t = threadIdx.x;                          // 128 threads: t = h*64+k
  int h = t >> 6, k = t & 63;
  float s = 0.0f;
  for (int c = 0; c < 64; ++c) s += Wedge[(size_t)(h * 64 + c) * 64 + k] * att[h * 64 + c];
  q[t] = s;
  __syncthreads();
  if (t < 2) {
    float lh = 0.0f, lt = 0.0f;
    for (int k2 = 0; k2 < 64; ++k2) {
      float qq = q[t * 64 + k2];
      lh += csh[k2] * Einv * qq;
      lt += cst[k2] * Einv * qq;
    }
    loop_h[t] = lh; loop_t[t] = lt;
  }
}

// per-node attention dot: out[n,h] = sum_c hs[n,h*C+c]*att[h,c]
__global__ void k_node_att(const float* __restrict__ hs, const float* __restrict__ att,
                           float* __restrict__ out, int N, int H, int C) {
  int n = blockIdx.x * blockDim.x + threadIdx.x;
  if (n >= N) return;
  int D = H * C;
  for (int h = 0; h < H; ++h) {
    float s = 0.0f;
    for (int c = 0; c < C; ++c) s += hs[(size_t)n * D + h * C + c] * att[h * C + c];
    out[n * H + h] = s;
  }
}

// edge softmax pass 1: numerator exp(leaky(logit)) and per-dst denominator (H=2)
__global__ void k_edge_pass1(const int* __restrict__ src, const int* __restrict__ dst,
                             int E, int Nloop,
                             const float* __restrict__ as_, const float* __restrict__ ad_,
                             const float* __restrict__ ea, const float* __restrict__ q,
                             const float* __restrict__ loopc,
                             float* __restrict__ num, float* __restrict__ denom, int Fea) {
  int e = blockIdx.x * blockDim.x + threadIdx.x;
  if (e >= E + Nloop) return;
  int s, d;
  float ae0 = 0.0f, ae1 = 0.0f;
  if (e < E) {
    s = src[e]; d = dst[e];
    if (ea) {
      const float* er = ea + (size_t)e * Fea;
      for (int k = 0; k < Fea; ++k) { float v = er[k]; ae0 += v * q[k]; ae1 += v * q[Fea + k]; }
    }
  } else {
    s = d = e - E;
    if (loopc) { ae0 = loopc[0]; ae1 = loopc[1]; }
  }
#pragma unroll
  for (int h = 0; h < 2; ++h) {
    float l = as_[s * 2 + h] + ad_[d * 2 + h] + (h ? ae1 : ae0);
    l = l > 0.0f ? l : 0.2f * l;                // leaky_relu 0.2
    float ex = __expf(l);                       // softmax is shift-invariant; skip max
    num[(size_t)e * 2 + h] = ex;
    atomicAdd(&denom[d * 2 + h], ex);
  }
}

// edge softmax pass 2: scatter hs[src]*alpha into out[dst, obase:obase+D]
__global__ void k_edge_pass2(const int* __restrict__ src, const int* __restrict__ dst,
                             int E, int Nloop, const float* __restrict__ hs, int D,
                             const float* __restrict__ num, const float* __restrict__ denom,
                             float* __restrict__ out, int ostride, int obase) {
  long long idx = (long long)blockIdx.x * blockDim.x + threadIdx.x;
  long long total = (long long)(E + Nloop) * D;
  if (idx >= total) return;
  int e = (int)(idx / D);
  int c = (int)(idx - (long long)e * D);
  int s, d;
  if (e < E) { s = src[e]; d = dst[e]; } else { s = d = e - E; }
  int h = (c * 2) / D;                          // head = c / (D/2)
  float alpha = num[(size_t)e * 2 + h] / (denom[d * 2 + h] + 1e-16f);
  atomicAdd(&out[(size_t)d * ostride + obase + c], hs[(size_t)s * D + c] * alpha);
}

// y = (f16) elu(x + bias[c])
__global__ void k_elu16(const float* __restrict__ x, const float* __restrict__ bias,
                        _Float16* __restrict__ y, long long total, int D) {
  long long idx = (long long)blockIdx.x * blockDim.x + threadIdx.x;
  if (idx >= total) return;
  int c = (int)(idx % D);
  float v = x[idx] + bias[c];
  v = v > 0.0f ? v : (__expf(v) - 1.0f);
  y[idx] = (_Float16)v;
}

__global__ void k_rep_bias(float* __restrict__ rep, const float* __restrict__ b0,
                           const float* __restrict__ b1, long long total) {
  long long idx = (long long)blockIdx.x * blockDim.x + threadIdx.x;
  if (idx >= total) return;
  int c = (int)(idx & 127);
  rep[idx] += (c < 64) ? b0[c] : b1[c - 64];
}

__global__ void k_pool_scatter(const float* __restrict__ v, const int* __restrict__ eb,
                               float* __restrict__ pool, int E) {
  long long idx = (long long)blockIdx.x * blockDim.x + threadIdx.x;
  if (idx >= (long long)E * 64) return;
  int e = (int)(idx >> 6); int c = (int)(idx & 63);
  atomicAdd(&pool[(size_t)eb[e] * 64 + c], v[idx]);
}

__global__ void k_sag_agg(const int* __restrict__ src, const int* __restrict__ dst,
                          const float* __restrict__ rep, float* __restrict__ agg, int E) {
  long long idx = (long long)blockIdx.x * blockDim.x + threadIdx.x;
  if (idx >= (long long)E * 128) return;
  int e = (int)(idx >> 7); int c = (int)(idx & 127);
  atomicAdd(&agg[(size_t)dst[e] * 128 + c], rep[(size_t)src[e] * 128 + c]);
}

__global__ void k_sag_score(const float* __restrict__ agg, const float* __restrict__ rep,
                            const float* __restrict__ Wrel, const float* __restrict__ brel,
                            const float* __restrict__ Wroot, const int* __restrict__ batch,
                            float* __restrict__ en, float* __restrict__ ssum, int N) {
  int n = blockIdx.x * blockDim.x + threadIdx.x;
  if (n >= N) return;
  float s = brel[0];
  for (int c = 0; c < 128; ++c)
    s += agg[(size_t)n * 128 + c] * Wrel[c] + rep[(size_t)n * 128 + c] * Wroot[c];
  float e = __expf(s);
  en[n] = e;
  atomicAdd(&ssum[batch[n]], e);
}

__global__ void k_sag_pool(const float* __restrict__ rep, const int* __restrict__ batch,
                           const float* __restrict__ en, const float* __restrict__ ssum,
                           float* __restrict__ outg, int N) {
  long long idx = (long long)blockIdx.x * blockDim.x + threadIdx.x;
  if (idx >= (long long)N * 128) return;
  int n = (int)(idx >> 7); int c = (int)(idx & 127);
  int g = batch[n];
  float w = en[n] / (ssum[g] + 1e-16f);
  atomicAdd(&outg[(size_t)g * 128 + c], rep[idx] * w);
}

__global__ void k_ln_stats(const float* __restrict__ rep, const int* __restrict__ batch,
                           float* __restrict__ gsum, float* __restrict__ gsumsq,
                           float* __restrict__ gcnt, int N) {
  int n = blockIdx.x * blockDim.x + threadIdx.x;
  if (n >= N) return;
  float s1 = 0.0f, s2 = 0.0f;
  for (int c = 0; c < 128; ++c) { float v = rep[(size_t)n * 128 + c]; s1 += v; s2 += v * v; }
  int g = batch[n];
  atomicAdd(&gsum[g], s1); atomicAdd(&gsumsq[g], s2); atomicAdd(&gcnt[g], 1.0f);
}

__global__ void k_ln_fin(const float* __restrict__ gsum, const float* __restrict__ gsumsq,
                         const float* __restrict__ gcnt, float* __restrict__ gmean,
                         float* __restrict__ grstd, int G) {
  int g = blockIdx.x * blockDim.x + threadIdx.x;
  if (g >= G) return;
  float cnt = gcnt[g]; if (cnt < 1.0f) cnt = 1.0f;
  float norm = cnt * 128.0f;
  float mean = gsum[g] / norm;
  float var  = gsumsq[g] / norm - mean * mean;
  if (var < 0.0f) var = 0.0f;
  gmean[g] = mean;
  grstd[g] = rsqrtf(var + 1e-5f);
}

__global__ void k_ln_apply(const float* __restrict__ rep, const int* __restrict__ batch,
                           const float* __restrict__ gmean, const float* __restrict__ grstd,
                           const float* __restrict__ w, const float* __restrict__ b,
                           float* __restrict__ out, int N) {
  long long idx = (long long)blockIdx.x * blockDim.x + threadIdx.x;
  if (idx >= (long long)N * 128) return;
  int n = (int)(idx >> 7); int c = (int)(idx & 127);
  int g = batch[n];
  float v = (rep[idx] - gmean[g]) * grstd[g] * w[c] + b[c];
  out[idx] = v > 0.0f ? v : 0.0f;
}

// ---------------------------------------------------------------------------
// host orchestration
// ---------------------------------------------------------------------------
extern "C" void kernel_launch(void* const* d_in, const int* in_sizes, int n_in,
                              void* d_out, int out_size, void* d_ws, size_t ws_size,
                              hipStream_t stream) {
  (void)n_in; (void)ws_size;
  const float* x_h        = (const float*)d_in[0];
  const float* x_t        = (const float*)d_in[1];
  const float* ea_h       = (const float*)d_in[2];
  const float* ea_t       = (const float*)d_in[3];
  const int*   ei_h       = (const int*)d_in[4];
  const int*   ei_t       = (const int*)d_in[5];
  const int*   bei        = (const int*)d_in[6];
  const int*   batch_h    = (const int*)d_in[7];
  const int*   batch_t    = (const int*)d_in[8];
  const int*   ebatch_h   = (const int*)d_in[9];
  const int*   ebatch_t   = (const int*)d_in[10];
  const float* W_fc       = (const float*)d_in[11];
  const float* att_src_fc = (const float*)d_in[12];
  const float* att_dst_fc = (const float*)d_in[13];
  const float* W_edge_fc  = (const float*)d_in[14];
  const float* att_edge   = (const float*)d_in[15];
  const float* b_fc       = (const float*)d_in[16];
  const float* W_lu       = (const float*)d_in[17];
  const float* b_lu       = (const float*)d_in[18];
  const float* W_intra    = (const float*)d_in[19];
  const float* att_src_i  = (const float*)d_in[20];
  const float* att_dst_i  = (const float*)d_in[21];
  const float* b_intra    = (const float*)d_in[22];
  const float* W_int_src  = (const float*)d_in[23];
  const float* W_int_dst  = (const float*)d_in[24];
  const float* att_src_b  = (const float*)d_in[25];
  const float* att_dst_b  = (const float*)d_in[26];
  const float* b_inter    = (const float*)d_in[27];
  const float* W_rel      = (const float*)d_in[28];
  const float* b_rel      = (const float*)d_in[29];
  const float* W_root     = (const float*)d_in[30];
  const float* ln_w       = (const float*)d_in[31];
  const float* ln_b       = (const float*)d_in[32];
  const float* W_re       = (const float*)d_in[33];
  const float* b_re       = (const float*)d_in[34];

  const int nN  = in_sizes[0] / 64;
  const int nE  = in_sizes[2] / 64;
  const int nEB = in_sizes[6] / 2;
  const int nG  = (int)(((long long)out_size - 2LL * nN * 128 - 2LL * nE * 64) / 256);

  const int* eih_s = ei_h;       const int* eih_d = ei_h + nE;
  const int* eit_s = ei_t;       const int* eit_d = ei_t + nE;
  const int* b_s   = bei;        const int* b_d   = bei + nEB;

  float* out_hx  = (float*)d_out;
  float* out_eah = out_hx  + (size_t)nN * 128;
  float* out_tx  = out_eah + (size_t)nE * 64;
  float* out_eat = out_tx  + (size_t)nN * 128;
  float* out_hg  = out_eat + (size_t)nE * 64;
  float* out_tg  = out_hg  + (size_t)nG * 128;

  // ---- workspace arena ----
  char* base = (char*)d_ws;
  size_t off = 0;
  auto alloc = [&](size_t bytes) -> void* {
    void* p = base + off;
    off += (bytes + 255) & ~(size_t)255;
    return p;
  };
  _Float16* Wfc16    = (_Float16*)alloc((size_t)128 * 64 * 2);
  _Float16* Wlu16    = (_Float16*)alloc((size_t)64 * 64 * 2);
  _Float16* Wintra16 = (_Float16*)alloc((size_t)64 * 128 * 2);
  _Float16* Wis16    = (_Float16*)alloc((size_t)64 * 128 * 2);
  _Float16* Wid16    = (_Float16*)alloc((size_t)64 * 128 * 2);
  _Float16* Wre16    = (_Float16*)alloc((size_t)128 * 64 * 2);
  float* q        = (float*)alloc(128 * 4);
  float* loop_h   = (float*)alloc(2 * 4);
  float* loop_t   = (float*)alloc(2 * 4);
  float* colsum_h = (float*)alloc(64 * 4);
  float* colsum_t = (float*)alloc(64 * 4);
  float* asbuf    = (float*)alloc((size_t)nN * 2 * 4);
  float* adbuf    = (float*)alloc((size_t)nN * 2 * 4);
  float* denom    = (float*)alloc((size_t)nN * 2 * 4);
  int maxE = nE > nEB ? nE : nEB;
  float* num      = (float*)alloc((size_t)(maxE + nN) * 2 * 4);
  float* en       = (float*)alloc((size_t)nN * 4);
  float* ssum     = (float*)alloc((size_t)nG * 4);
  float* gsum     = (float*)alloc((size_t)nG * 4);
  float* gsumsq   = (float*)alloc((size_t)nG * 4);
  float* gcnt     = (float*)alloc((size_t)nG * 4);
  float* gmean    = (float*)alloc((size_t)nG * 4);
  float* grstd    = (float*)alloc((size_t)nG * 4);
  float* pool_h   = (float*)alloc((size_t)nG * 64 * 4);
  float* pool_t   = (float*)alloc((size_t)nG * 64 * 4);
  _Float16* pool16 = (_Float16*)alloc((size_t)nG * 64 * 2);
  _Float16* eh16   = (_Float16*)alloc((size_t)nN * 128 * 2);
  _Float16* et16   = (_Float16*)alloc((size_t)nN * 128 * 2);
  _Float16* act16  = (_Float16*)alloc((size_t)nE * 64 * 2);          // x / ea staging
  float* big       = (float*)alloc((size_t)nE * 64 * 4);             // >= max(N*128, 2*N*64)
  float* rep_h     = (float*)alloc((size_t)nN * 128 * 4);
  float* rep_t     = (float*)alloc((size_t)nN * 128 * 4);

  const int TB = 256;
  auto nb = [](long long n, int bs) { return (unsigned)((n + bs - 1) / bs); };
  auto zero = [&](float* p, long long n) {
    unsigned b = nb(n, TB); if (b > 4096u) b = 4096u;
    k_fill_zero<<<dim3(b), dim3(TB), 0, stream>>>(p, n);
  };
  auto cvt = [&](const float* x, _Float16* y, long long n) {
    unsigned b = nb(n, TB); if (b > 8192u) b = 8192u;
    k_cvt_f16<<<dim3(b), dim3(TB), 0, stream>>>(x, y, n);
  };
  auto gemmB = [&](int M) {
    unsigned waves = (unsigned)((M + 15) / 16);
    return dim3((waves * 32 + TB - 1) / TB);
  };

  // ---- stage 0: weight conversion, edge-attention query vectors ----
  cvt(W_fc, Wfc16, 128 * 64);
  cvt(W_lu, Wlu16, 64 * 64);
  cvt(W_intra, Wintra16, 64 * 128);
  cvt(W_int_src, Wis16, 64 * 128);
  cvt(W_int_dst, Wid16, 64 * 128);
  cvt(W_re, Wre16, 128 * 64);
  zero(colsum_h, 64); zero(colsum_t, 64);
  k_colsum64<<<dim3((unsigned)((nE + 2047) / 2048)), dim3(64), 0, stream>>>(ea_h, nE, colsum_h);
  k_colsum64<<<dim3((unsigned)((nE + 2047) / 2048)), dim3(64), 0, stream>>>(ea_t, nE, colsum_t);
  k_qprep<<<dim3(1), dim3(128), 0, stream>>>(W_edge_fc, att_edge, colsum_h, colsum_t,
                                             1.0f / (float)nE, q, loop_h, loop_t);

  // ---- stage 1: feature GATConv (per side), producing eh16/et16 ----
  auto feature_gat = [&](const float* x, const float* ea, const int* es, const int* ed,
                         const float* loopc, float* aggrep, _Float16* out16) {
    cvt(x, act16, (long long)nN * 64);
    k_gemm_wmma<64, 128, 0><<<gemmB(nN), dim3(TB), 0, stream>>>(act16, Wfc16, nullptr, big, nullptr, nN);
    k_node_att<<<dim3(nb(nN, TB)), dim3(TB), 0, stream>>>(big, att_src_fc, asbuf, nN, 2, 64);
    k_node_att<<<dim3(nb(nN, TB)), dim3(TB), 0, stream>>>(big, att_dst_fc, adbuf, nN, 2, 64);
    zero(denom, (long long)nN * 2);
    k_edge_pass1<<<dim3(nb((long long)nE + nN, TB)), dim3(TB), 0, stream>>>(
        es, ed, nE, nN, asbuf, adbuf, ea, q, loopc, num, denom, 64);
    zero(aggrep, (long long)nN * 128);
    k_edge_pass2<<<dim3(nb((long long)(nE + nN) * 128, TB)), dim3(TB), 0, stream>>>(
        es, ed, nE, nN, big, 128, num, denom, aggrep, 128, 0);
    k_elu16<<<dim3(nb((long long)nN * 128, TB)), dim3(TB), 0, stream>>>(
        aggrep, b_fc, out16, (long long)nN * 128, 128);
  };
  feature_gat(x_h, ea_h, eih_s, eih_d, loop_h, rep_h, eh16);
  feature_gat(x_t, ea_t, eit_s, eit_d, loop_t, rep_t, et16);

  // ---- stage 2: ea' = ea @ W_lu^T + b_lu ; relu -> out ; graph pooling ----
  auto edge_linear = [&](const float* ea, const int* eb, float* out_relu, float* pool) {
    cvt(ea, act16, (long long)nE * 64);
    k_gemm_wmma<64, 64, 1><<<gemmB(nE), dim3(TB), 0, stream>>>(act16, Wlu16, b_lu, big, out_relu, nE);
    zero(pool, (long long)nG * 64);
    k_pool_scatter<<<dim3(nb((long long)nE * 64, TB)), dim3(TB), 0, stream>>>(big, eb, pool, nE);
  };
  edge_linear(ea_h, ebatch_h, out_eah, pool_h);
  edge_linear(ea_t, ebatch_t, out_eat, pool_t);

  // ---- stage 3: intra + inter GATs into rep_h / rep_t ----
  zero(rep_h, (long long)nN * 128);
  zero(rep_t, (long long)nN * 128);
  auto gat64 = [&](const _Float16* s16, const _Float16* d16,
                   const float* attS, const float* attD,
                   const _Float16* Ws16, const _Float16* Wd16,
                   const int* es, const int* ed, int Ereal, float* rep, int obase) {
    float* hsA = big;
    float* hdB = big + (size_t)nN * 64;
    k_gemm_wmma<128, 64, 0><<<gemmB(nN), dim3(TB), 0, stream>>>(s16, Ws16, nullptr, hsA, nullptr, nN);
    if (s16 == d16 && Ws16 == Wd16) hdB = hsA;
    else k_gemm_wmma<128, 64, 0><<<gemmB(nN), dim3(TB), 0, stream>>>(d16, Wd16, nullptr, hdB, nullptr, nN);
    k_node_att<<<dim3(nb(nN, TB)), dim3(TB), 0, stream>>>(hsA, attS, asbuf, nN, 2, 32);
    k_node_att<<<dim3(nb(nN, TB)), dim3(TB), 0, stream>>>(hdB, attD, adbuf, nN, 2, 32);
    zero(denom, (long long)nN * 2);
    k_edge_pass1<<<dim3(nb((long long)Ereal + nN, TB)), dim3(TB), 0, stream>>>(
        es, ed, Ereal, nN, asbuf, adbuf, nullptr, nullptr, nullptr, num, denom, 0);
    k_edge_pass2<<<dim3(nb((long long)(Ereal + nN) * 64, TB)), dim3(TB), 0, stream>>>(
        es, ed, Ereal, nN, hsA, 64, num, denom, rep, 128, obase);
  };
  gat64(eh16, eh16, att_src_i, att_dst_i, Wintra16, Wintra16, eih_s, eih_d, nE, rep_h, 0);
  gat64(et16, et16, att_src_i, att_dst_i, Wintra16, Wintra16, eit_s, eit_d, nE, rep_t, 0);
  gat64(eh16, et16, att_src_b, att_dst_b, Wis16, Wid16, b_s, b_d, nEB, rep_t, 64);   // t_inter
  gat64(et16, eh16, att_src_b, att_dst_b, Wis16, Wid16, b_d, b_s, nEB, rep_h, 64);   // h_inter
  k_rep_bias<<<dim3(nb((long long)nN * 128, TB)), dim3(TB), 0, stream>>>(rep_h, b_intra, b_inter, (long long)nN * 128);
  k_rep_bias<<<dim3(nb((long long)nN * 128, TB)), dim3(TB), 0, stream>>>(rep_t, b_intra, b_inter, (long long)nN * 128);

  // ---- stage 4: SAG readout + pooled-edge GEMM fused into graph outputs ----
  auto sag = [&](const float* rep, const int* es, const int* ed, const int* batch,
                 const float* pool, float* outg) {
    zero(big, (long long)nN * 128);
    k_sag_agg<<<dim3(nb((long long)nE * 128, TB)), dim3(TB), 0, stream>>>(es, ed, rep, big, nE);
    zero(ssum, nG);
    k_sag_score<<<dim3(nb(nN, TB)), dim3(TB), 0, stream>>>(big, rep, W_rel, b_rel, W_root, batch, en, ssum, nN);
    zero(outg, (long long)nG * 128);
    k_sag_pool<<<dim3(nb((long long)nN * 128, TB)), dim3(TB), 0, stream>>>(rep, batch, en, ssum, outg, nN);
    cvt(pool, pool16, (long long)nG * 64);
    k_gemm_wmma<64, 128, 2><<<gemmB(nG), dim3(TB), 0, stream>>>(pool16, Wre16, b_re, outg, nullptr, nG);
  };
  sag(rep_h, eih_s, eih_d, batch_h, pool_h, out_hg);
  sag(rep_t, eit_s, eit_d, batch_t, pool_t, out_tg);

  // ---- stage 5: graph layer-norm + relu -> node outputs ----
  auto ln = [&](const float* rep, const int* batch, float* outx) {
    zero(gsum, nG); zero(gsumsq, nG); zero(gcnt, nG);
    k_ln_stats<<<dim3(nb(nN, TB)), dim3(TB), 0, stream>>>(rep, batch, gsum, gsumsq, gcnt, nN);
    k_ln_fin<<<dim3(nb(nG, TB)), dim3(TB), 0, stream>>>(gsum, gsumsq, gcnt, gmean, grstd, nG);
    k_ln_apply<<<dim3(nb((long long)nN * 128, TB)), dim3(TB), 0, stream>>>(
        rep, batch, gmean, grstd, ln_w, ln_b, outx, nN);
  };
  ln(rep_h, batch_h, out_hx);
  ln(rep_t, batch_t, out_tx);
}
